// ConfusionPointerNetVectorized_85856396247702
// MI455X (gfx1250) — compile-verified, matrix operand order checked by reading, not run on hardware
//
#include <hip/hip_runtime.h>
#include <hip/hip_bf16.h>
#include <math.h>

typedef __bf16 bf16;
typedef __attribute__((ext_vector_type(16))) __bf16 bf16x16;
typedef __attribute__((ext_vector_type(8)))  __bf16 bf16x8;
typedef __attribute__((ext_vector_type(8)))  float  v8f;

#define NEGINF (-1000000000.0f)

// ---- dims ----
#define BV 8000
#define BE 384
#define BH 512
#define BD 512
#define BA 512
#define BP 128
#define BB 16
#define BN 128
#define G4H 2048   // 4*H == 4*D

__device__ __forceinline__ float sigf(float x) { return 1.0f / (1.0f + __expf(-x)); }

// Load one 16x32 bf16 fragment half-row: two contiguous 16B loads.
// p must be 16B aligned (all strides are multiples of 8 elements).
__device__ __forceinline__ bf16x16 load_frag(const bf16* __restrict__ p) {
    bf16x8 lo = *(const bf16x8*)p;
    bf16x8 hi = *(const bf16x8*)(p + 16);
    bf16x16 r;
    #pragma unroll
    for (int e = 0; e < 8; ++e) { r[e] = lo[e]; r[8 + e] = hi[e]; }
    return r;
}

// ============================================================================
// Wave-tile WMMA GEMM (single code path):  C = epilogue( A @ W^T + bias )
//   A: (M,K) bf16 row-major, lda    W: (Nout,K) bf16 row-major, ldb
//   EPI: 0=none, 1=tanh, 2=confusion-mask (int mask, same shape as C)
//   batched over blockIdx.z with element strides aB/bB/cB.
// Fragment packing per ISA 16-bit layouts: lane half 'hi' holds
// K = hi*8+0..7 and 16+hi*8+0..7 of row (row0+l) / weight-row (col0+l).
// ============================================================================
template <int EPI>
__global__ void wmma_gemm_w(
    const bf16* __restrict__ A, long long aB, int lda,
    const bf16* __restrict__ W, long long bB, int ldb,
    const float* __restrict__ bias,
    float* __restrict__ C, long long cB, int ldc,
    const int* __restrict__ mask,
    int M, int Nout, int K)
{
    const int wave = threadIdx.x >> 5;
    const int lane = threadIdx.x & 31;
    const int l  = lane & 15;
    const int hi = lane >> 4;

    const int tilesN = Nout >> 4;
    const int tilesM = M >> 4;
    long long tile = (long long)blockIdx.x * (blockDim.x >> 5) + wave;
    if (tile >= (long long)tilesM * tilesN) return;   // uniform per-wave
    const int tm = (int)(tile / tilesN);
    const int tn = (int)(tile % tilesN);
    const int row0 = tm << 4, col0 = tn << 4;

    const bf16* __restrict__ ap = A + (long long)blockIdx.z * aB
                                    + (long long)(row0 + l) * lda + hi * 8;
    const bf16* __restrict__ bp = W + (long long)blockIdx.z * bB
                                    + (long long)(col0 + l) * ldb + hi * 8;
    float* Cb = C + (long long)blockIdx.z * cB;

    v8f acc = {};
    for (int k0 = 0; k0 < K; k0 += 32, ap += 32, bp += 32) {
        __builtin_prefetch(ap + 64, 0, 1);   // next cacheline of A stream
        __builtin_prefetch(bp + 64, 0, 1);   // next cacheline of W stream
        bf16x16 a = load_frag(ap);
        bf16x16 b = load_frag(bp);
        acc = __builtin_amdgcn_wmma_f32_16x16x32_bf16(
                  false, a, false, b, (short)0, acc, false, false);
    }

    const float bv = bias ? bias[col0 + l] : 0.0f;
    #pragma unroll
    for (int v = 0; v < 8; ++v) {
        const int r = row0 + v + hi * 8;           // M index
        long long idx = (long long)r * ldc + col0 + l;
        float val = acc[v] + bv;
        if (EPI == 1) {
            val = tanhf(val);
        } else if (EPI == 2) {
            if (mask[idx] == 0) val = NEGINF;
        }
        Cb[idx] = val;
    }
}

// ============================================================================
// Fused triple-LSTM recurrence. blockIdx.x selects {enc_fwd, enc_bwd, dec}.
// One persistent 1024-thread workgroup per LSTM. h/c/gates live in LDS:
//   gates: 16x2048 f32 (128KB) | c: 16x512 f32 (32KB) | h(bf16): 16x512 (16KB)
// Per step: 128 WMMA output tiles (M=batch=16, Nout=2048, K=512), 16 k-steps.
// ============================================================================
__global__ __launch_bounds__(1024)
void lstm3_recurrence(
    const float* Xp0, const float* Xp1, const float* Xp2,   // (B,N,2048) each
    const bf16*  W0,  const bf16*  W1w, const bf16*  W2w,   // Whh bf16 (2048,512)
    float* H0, float* H1, float* H2,                        // outputs
    int ld0, int ld1, int ld2, int off0, int off1, int off2,
    int rev0, int rev1, int rev2)
{
    const float* Xproj = (blockIdx.x == 0) ? Xp0 : (blockIdx.x == 1) ? Xp1 : Xp2;
    const bf16*  Whh   = (blockIdx.x == 0) ? W0  : (blockIdx.x == 1) ? W1w : W2w;
    float*       Hout  = (blockIdx.x == 0) ? H0  : (blockIdx.x == 1) ? H1  : H2;
    const int outLd  = (blockIdx.x == 0) ? ld0  : (blockIdx.x == 1) ? ld1  : ld2;
    const int outOff = (blockIdx.x == 0) ? off0 : (blockIdx.x == 1) ? off1 : off2;
    const int rev    = (blockIdx.x == 0) ? rev0 : (blockIdx.x == 1) ? rev1 : rev2;

    extern __shared__ char smem[];
    float* gates = (float*)smem;                      // 16 * 2048
    float* cst   = gates + BB * G4H;                  // 16 * 512
    bf16*  hbf   = (bf16*)(cst + BB * BH);            // 16 * 512

    const int tid = threadIdx.x;
    for (int i = tid; i < BB * BH; i += blockDim.x) { cst[i] = 0.0f; hbf[i] = (bf16)0.0f; }
    __syncthreads();

    const int wave = tid >> 5, lane = tid & 31;
    const int l = lane & 15, hi = lane >> 4;
    const bf16* hrow = hbf + l * BH + hi * 8;         // A operand row (batch l)

    for (int step = 0; step < BN; ++step) {
        const int ts = rev ? (BN - 1 - step) : step;

        // ---- gate GEMM: gates = Xproj[:,ts,:] + h @ Whh^T ----
        #pragma unroll
        for (int i = 0; i < 4; ++i) {
            const int col0 = (wave * 4 + i) << 4;
            v8f acc;
            #pragma unroll
            for (int v = 0; v < 8; ++v)
                acc[v] = Xproj[((long long)(v + hi * 8) * BN + ts) * G4H + col0 + l];
            const bf16* ap = hrow;
            const bf16* bp = Whh + (long long)(col0 + l) * BH + hi * 8;
            for (int k0 = 0; k0 < BH; k0 += 32, ap += 32, bp += 32) {
                bf16x16 a = load_frag(ap);          // ds_load_b128 x2
                bf16x16 b = load_frag(bp);
                acc = __builtin_amdgcn_wmma_f32_16x16x32_bf16(
                          false, a, false, b, (short)0, acc, false, false);
            }
            #pragma unroll
            for (int v = 0; v < 8; ++v)
                gates[(v + hi * 8) * G4H + col0 + l] = acc[v];
        }
        __syncthreads();

        // ---- pointwise LSTM cell ----
        for (int idx = tid; idx < BB * BH; idx += blockDim.x) {
            const int bb = idx >> 9, j = idx & 511;
            const float* g = gates + bb * G4H;
            const float ig = sigf(g[j]);
            const float fg = sigf(g[BH + j]);
            const float gg = tanhf(g[2 * BH + j]);
            const float og = sigf(g[3 * BH + j]);
            const float c  = fg * cst[idx] + ig * gg;
            cst[idx] = c;
            const float h = og * tanhf(c);
            hbf[idx] = (bf16)h;
            Hout[((long long)bb * BN + ts) * outLd + outOff + j] = h;
        }
        __syncthreads();
    }
}

// ---- small helper kernels --------------------------------------------------

__global__ void f32_to_bf16_k(const float* __restrict__ in, bf16* __restrict__ out, long long n) {
    long long i = (long long)blockIdx.x * blockDim.x + threadIdx.x;
    if (i < n) out[i] = (bf16)in[i];
}

__global__ void embed_gather_k(const int* __restrict__ ids, const float* __restrict__ emb,
                               bf16* __restrict__ xbf) {
    long long i = (long long)blockIdx.x * blockDim.x + threadIdx.x;
    if (i >= (long long)BB * BN * BE) return;
    const int r = (int)(i / BE), e = (int)(i % BE);
    xbf[i] = (bf16)emb[(long long)ids[r] * BE + e];
}

// enc_hsT[b][c][k] = enc_hs[b][k][c]   (weight layout for the context GEMM)
__global__ void enc_transpose_k(const float* __restrict__ enc, bf16* __restrict__ encT) {
    long long i = (long long)blockIdx.x * blockDim.x + threadIdx.x;
    const long long total = (long long)BB * 2 * BH * BN;
    if (i >= total) return;
    const int b   = (int)(i / (2 * BH * BN));
    const int rem = (int)(i % (2 * BH * BN));
    const int c = rem / BN, k = rem % BN;
    encT[i] = (bf16)enc[((long long)b * BN + k) * (2 * BH) + c];
}

// dense repack of ptr_W[:, :512] (row stride 513 -> 512, f32 -> bf16)
__global__ void ptrw_repack_k(const float* __restrict__ ptrW, bf16* __restrict__ outw) {
    long long i = (long long)blockIdx.x * blockDim.x + threadIdx.x;
    if (i >= (long long)BP * BD) return;
    const int p = (int)(i / BD), k = (int)(i % BD);
    outw[i] = (bf16)ptrW[(long long)p * (BD + 1) + k];
}

// score[b,t,n] = sum_a tanh(dterm[b,t,a] + eterm[b,n,a]) * v_att[a]  (+ mask)
__global__ void attn_score_k(const float* __restrict__ dterm, const float* __restrict__ eterm,
                             const float* __restrict__ v_att, const int* __restrict__ src_mask,
                             float* __restrict__ score) {
    const int bt = blockIdx.x, b = bt >> 7;
    const int n  = threadIdx.x;
    __shared__ float dsh[BA];
    __shared__ float vsh[BA];
    for (int a = threadIdx.x; a < BA; a += blockDim.x) {
        dsh[a] = dterm[(long long)bt * BA + a];
        vsh[a] = v_att[a];
    }
    __syncthreads();
    const float* e = eterm + ((long long)b * BN + n) * BA;
    float s = 0.0f;
    for (int a = 0; a < BA; ++a) s += tanhf(dsh[a] + e[a]) * vsh[a];
    if (src_mask[b * BN + n] == 0) s = NEGINF;
    score[(long long)bt * BN + n] = s;
}

__global__ void softmax_row_k(const float* __restrict__ score, bf16* __restrict__ alpha_bf) {
    const int bt = blockIdx.x, n = threadIdx.x;
    __shared__ float sh[BN];
    const float v = score[(long long)bt * BN + n];
    sh[n] = v; __syncthreads();
    for (int off = 64; off > 0; off >>= 1) { if (n < off) sh[n] = fmaxf(sh[n], sh[n + off]); __syncthreads(); }
    const float mx = sh[0]; __syncthreads();
    const float ex = __expf(v - mx);
    sh[n] = ex; __syncthreads();
    for (int off = 64; off > 0; off >>= 1) { if (n < off) sh[n] += sh[n + off]; __syncthreads(); }
    alpha_bf[(long long)bt * BN + n] = (bf16)(ex / sh[0]);
}

__global__ void build_concat_k(const float* __restrict__ dec_h, const float* __restrict__ ctx,
                               bf16* __restrict__ out) {
    long long i = (long long)blockIdx.x * blockDim.x + threadIdx.x;
    const long long total = (long long)BB * BN * (BD + 2 * BH);
    if (i >= total) return;
    const int cols = BD + 2 * BH;                 // 1536
    const int r = (int)(i / cols), c = (int)(i % cols);
    const float v = (c < BD) ? dec_h[(long long)r * BD + c]
                             : ctx[(long long)r * (2 * BH) + (c - BD)];
    out[i] = (bf16)v;
}

__global__ void row_sum_512_k(const float* __restrict__ in, float* __restrict__ out) {
    __shared__ float sh[256];
    const int r = blockIdx.x;
    float s = in[(long long)r * 512 + threadIdx.x] + in[(long long)r * 512 + threadIdx.x + 256];
    sh[threadIdx.x] = s; __syncthreads();
    for (int off = 128; off > 0; off >>= 1) { if (threadIdx.x < off) sh[threadIdx.x] += sh[threadIdx.x + off]; __syncthreads(); }
    if (threadIdx.x == 0) out[r] = sh[0];
}

// pointer head: loc = eye(N) => pos_scores[b,t,n] has only 2 distinct values
__global__ void pointer_scores_k(const float* __restrict__ base, const float* __restrict__ ptrW,
                                 const float* __restrict__ ps_W, const float* __restrict__ ps_b,
                                 const float* __restrict__ sentinel, const int* __restrict__ src_mask,
                                 float* __restrict__ out) {
    const int bt = blockIdx.x, b = bt >> 7, t = bt & 127;
    const int p  = threadIdx.x;      // 128
    __shared__ float s0sh[BP];
    __shared__ float s1sh[BP];
    const float v  = base[(long long)bt * BP + p];
    const float w  = ps_W[p];
    const float pw = ptrW[p * (BD + 1) + BD];
    s0sh[p] = tanhf(v) * w;
    s1sh[p] = tanhf(v + pw) * w;
    __syncthreads();
    for (int off = 64; off > 0; off >>= 1) {
        if (p < off) { s0sh[p] += s0sh[p + off]; s1sh[p] += s1sh[p + off]; }
        __syncthreads();
    }
    const float s0 = s0sh[0] + ps_b[0];
    const float s1 = s1sh[0] + ps_b[0];
    const int n = p;
    float val = (n == t) ? s1 : s0;
    if (src_mask[b * BN + n] == 0) val = NEGINF;
    out[(long long)bt * (BN + 1) + n] = val;
    if (p == 0) out[(long long)bt * (BN + 1) + BN] = sentinel[bt];
}

// ---- host side -------------------------------------------------------------

static inline void launch_gemm(hipStream_t s,
    const bf16* A, long long aB, int lda,
    const bf16* W, long long bB, int ldb,
    const float* bias, float* C, long long cB, int ldc,
    const int* mask, int M, int Nout, int K, int epi, int batch)
{
    long long tiles = (long long)(M / 16) * (Nout / 16);
    int blocks = (int)((tiles + 7) / 8);
    dim3 grid(blocks, 1, batch);
    if (epi == 1)
        wmma_gemm_w<1><<<grid, 256, 0, s>>>(A, aB, lda, W, bB, ldb, bias, C, cB, ldc, mask, M, Nout, K);
    else if (epi == 2)
        wmma_gemm_w<2><<<grid, 256, 0, s>>>(A, aB, lda, W, bB, ldb, bias, C, cB, ldc, mask, M, Nout, K);
    else
        wmma_gemm_w<0><<<grid, 256, 0, s>>>(A, aB, lda, W, bB, ldb, bias, C, cB, ldc, mask, M, Nout, K);
}

static inline void launch_cvt(hipStream_t s, const float* in, bf16* out, long long n) {
    f32_to_bf16_k<<<(unsigned)((n + 255) / 256), 256, 0, s>>>(in, out, n);
}

extern "C" void kernel_launch(void* const* d_in, const int* in_sizes, int n_in,
                              void* d_out, int out_size, void* d_ws, size_t ws_size,
                              hipStream_t stream) {
    (void)in_sizes; (void)n_in; (void)out_size; (void)ws_size;

    const int*   src_ids  = (const int*)d_in[0];
    const int*   src_mask = (const int*)d_in[1];
    const int*   cmask    = (const int*)d_in[2];
    const float* emb      = (const float*)d_in[3];
    const float* Wih_f = (const float*)d_in[4];  const float* Whh_f = (const float*)d_in[5];  const float* b_f = (const float*)d_in[6];
    const float* Wih_b = (const float*)d_in[7];  const float* Whh_b = (const float*)d_in[8];  const float* b_b = (const float*)d_in[9];
    const float* Wih_d = (const float*)d_in[10]; const float* Whh_d = (const float*)d_in[11]; const float* b_d = (const float*)d_in[12];
    const float* W1 = (const float*)d_in[13];
    const float* W2 = (const float*)d_in[14];
    const float* v_att = (const float*)d_in[15];
    const float* combine_W = (const float*)d_in[16]; const float* combine_b = (const float*)d_in[17];
    const float* vocab_W   = (const float*)d_in[18]; const float* vocab_b   = (const float*)d_in[19];
    const float* Wg_W      = (const float*)d_in[20]; const float* Wg_b      = (const float*)d_in[21];
    const float* ptr_W     = (const float*)d_in[22]; const float* ptr_b     = (const float*)d_in[23];
    const float* ps_W      = (const float*)d_in[24]; const float* ps_b      = (const float*)d_in[25];

    float* out = (float*)d_out;
    const long long M = (long long)BB * BN;      // 2048 rows (b,t)

    // --- workspace carve ---
    char* ws = (char*)d_ws;
    size_t off = 0;
    auto alloc = [&](size_t bytes) -> void* {
        void* p = (void*)(ws + off);
        off += (bytes + 255) & ~(size_t)255;
        return p;
    };
    bf16* x_bf       = (bf16*)alloc(M * BE * 2);
    bf16* Wih_f_bf   = (bf16*)alloc((size_t)G4H * BE * 2);
    bf16* Wih_b_bf   = (bf16*)alloc((size_t)G4H * BE * 2);
    bf16* Wih_d_bf   = (bf16*)alloc((size_t)G4H * BE * 2);
    bf16* Whh_f_bf   = (bf16*)alloc((size_t)G4H * BH * 2);
    bf16* Whh_b_bf   = (bf16*)alloc((size_t)G4H * BH * 2);
    bf16* Whh_d_bf   = (bf16*)alloc((size_t)G4H * BD * 2);
    bf16* W1_bf      = (bf16*)alloc((size_t)BA * BD * 2);
    bf16* W2_bf      = (bf16*)alloc((size_t)BA * 2 * BH * 2);
    bf16* combine_bf = (bf16*)alloc((size_t)BD * (BD + 2 * BH) * 2);
    bf16* vocab_bf   = (bf16*)alloc((size_t)BV * BD * 2);
    bf16* Wg_bf      = (bf16*)alloc((size_t)BD * BD * 2);
    bf16* ptrW512_bf = (bf16*)alloc((size_t)BP * BD * 2);
    float* Xproj_f   = (float*)alloc(M * G4H * 4);
    float* Xproj_b   = (float*)alloc(M * G4H * 4);
    float* Xproj_d   = (float*)alloc(M * G4H * 4);
    float* enc_hs    = (float*)alloc(M * 2 * BH * 4);
    bf16*  enc_hs_bf = (bf16*)alloc(M * 2 * BH * 2);
    bf16*  enc_hsT_bf= (bf16*)alloc(M * 2 * BH * 2);   // (B, 2H, N) weight layout
    float* dec_h     = (float*)alloc(M * BD * 4);
    bf16*  dec_h_bf  = (bf16*)alloc(M * BD * 2);
    float* dterm     = (float*)alloc(M * BA * 4);
    float* eterm     = (float*)alloc(M * BA * 4);
    float* score     = (float*)alloc(M * BN * 4);
    bf16*  alpha_bf  = (bf16*)alloc(M * BN * 2);
    float* context   = (float*)alloc(M * 2 * BH * 4);
    bf16*  concat_bf = (bf16*)alloc(M * (BD + 2 * BH) * 2);
    float* combined  = (float*)alloc(M * BD * 4);
    bf16*  combined_bf = (bf16*)alloc(M * BD * 2);
    float* gtan      = (float*)alloc(M * BD * 4);
    float* sentinel  = (float*)alloc(M * 4);
    float* base      = (float*)alloc(M * BP * 4);

    // --- 1. weight conversions to bf16 ---
    launch_cvt(stream, Wih_f, Wih_f_bf, (long long)G4H * BE);
    launch_cvt(stream, Wih_b, Wih_b_bf, (long long)G4H * BE);
    launch_cvt(stream, Wih_d, Wih_d_bf, (long long)G4H * BE);
    launch_cvt(stream, Whh_f, Whh_f_bf, (long long)G4H * BH);
    launch_cvt(stream, Whh_b, Whh_b_bf, (long long)G4H * BH);
    launch_cvt(stream, Whh_d, Whh_d_bf, (long long)G4H * BD);
    launch_cvt(stream, W1, W1_bf, (long long)BA * BD);
    launch_cvt(stream, W2, W2_bf, (long long)BA * 2 * BH);
    launch_cvt(stream, combine_W, combine_bf, (long long)BD * (BD + 2 * BH));
    launch_cvt(stream, vocab_W, vocab_bf, (long long)BV * BD);
    launch_cvt(stream, Wg_W, Wg_bf, (long long)BD * BD);
    ptrw_repack_k<<<(unsigned)(((long long)BP * BD + 255) / 256), 256, 0, stream>>>(ptr_W, ptrW512_bf);

    // --- 2. embedding gather (f32 -> bf16) ---
    embed_gather_k<<<(unsigned)((M * BE + 255) / 256), 256, 0, stream>>>(src_ids, emb, x_bf);

    // --- 3. LSTM input projections, hoisted out of the time loop ---
    launch_gemm(stream, x_bf, 0, BE, Wih_f_bf, 0, BE, b_f, Xproj_f, 0, G4H, nullptr, (int)M, G4H, BE, 0, 1);
    launch_gemm(stream, x_bf, 0, BE, Wih_b_bf, 0, BE, b_b, Xproj_b, 0, G4H, nullptr, (int)M, G4H, BE, 0, 1);
    launch_gemm(stream, x_bf, 0, BE, Wih_d_bf, 0, BE, b_d, Xproj_d, 0, G4H, nullptr, (int)M, G4H, BE, 0, 1);

    // --- 4. all three recurrences in one kernel (3 persistent workgroups) ---
    const size_t lds = (size_t)BB * G4H * 4 + (size_t)BB * BH * 4 + (size_t)BB * BH * 2; // 176KB
    lstm3_recurrence<<<3, 1024, lds, stream>>>(
        Xproj_f, Xproj_b, Xproj_d,
        Whh_f_bf, Whh_b_bf, Whh_d_bf,
        enc_hs, enc_hs, dec_h,
        2 * BH, 2 * BH, BD,     // out ld
        0, BH, 0,               // out offset (bwd writes cols 512..1023)
        0, 1, 0);               // reverse flags

    // --- 5. activations to bf16 (+ per-batch transpose for the context GEMM) ---
    launch_cvt(stream, enc_hs, enc_hs_bf, M * 2 * BH);
    launch_cvt(stream, dec_h, dec_h_bf, M * BD);
    enc_transpose_k<<<(unsigned)((M * 2 * BH + 255) / 256), 256, 0, stream>>>(enc_hs, enc_hsT_bf);

    // --- 6. attention projections ---
    launch_gemm(stream, dec_h_bf, 0, BD, W1_bf, 0, BD, nullptr, dterm, 0, BA, nullptr, (int)M, BA, BD, 0, 1);
    launch_gemm(stream, enc_hs_bf, 0, 2 * BH, W2_bf, 0, 2 * BH, nullptr, eterm, 0, BA, nullptr, (int)M, BA, 2 * BH, 0, 1);

    // --- 7. additive-attention score + softmax ---
    attn_score_k<<<(unsigned)M, BN, 0, stream>>>(dterm, eterm, v_att, src_mask, score);
    softmax_row_k<<<(unsigned)M, BN, 0, stream>>>(score, alpha_bf);

    // --- 8. context = alpha @ enc_hs  (B uses pre-transposed weight layout) ---
    launch_gemm(stream, alpha_bf, (long long)BN * BN, BN,
                enc_hsT_bf, (long long)(2 * BH) * BN, BN,
                nullptr, context, (long long)BN * 2 * BH, 2 * BH, nullptr,
                BN, 2 * BH, BN, 0, BB);

    // --- 9. combined = tanh(concat(dec_h, context) @ combine_W^T + b) ---
    build_concat_k<<<(unsigned)((M * (BD + 2 * BH) + 255) / 256), 256, 0, stream>>>(dec_h, context, concat_bf);
    launch_gemm(stream, concat_bf, 0, BD + 2 * BH, combine_bf, 0, BD + 2 * BH, combine_b,
                combined, 0, BD, nullptr, (int)M, BD, BD + 2 * BH, 1, 1);
    launch_cvt(stream, combined, combined_bf, M * BD);

    // --- 10. vocab logits (masked), written straight into d_out tail ---
    float* vocab_out = out + M * (BN + 1);
    launch_gemm(stream, combined_bf, 0, BD, vocab_bf, 0, BD, vocab_b,
                vocab_out, 0, BV, cmask, (int)M, BV, BD, 2, 1);

    // --- 11. sentinel = sum(tanh(combined @ Wg^T + b)) ---
    launch_gemm(stream, combined_bf, 0, BD, Wg_bf, 0, BD, Wg_b,
                gtan, 0, BD, nullptr, (int)M, BD, BD, 1, 1);
    row_sum_512_k<<<(unsigned)M, 256, 0, stream>>>(gtan, sentinel);

    // --- 12. pointer base + assembled pointer logits ---
    launch_gemm(stream, combined_bf, 0, BD, ptrW512_bf, 0, BD, ptr_b,
                base, 0, BP, nullptr, (int)M, BP, BD, 0, 1);
    pointer_scores_k<<<(unsigned)M, BP, 0, stream>>>(base, ptr_W, ps_W, ps_b, sentinel, src_mask, out);
}
